// EncoderBlock_3753801417046
// MI455X (gfx1250) — matrix operand order repaired and checked by reading, not compiled, not run on hardware
//
#include <hip/hip_runtime.h>

typedef __attribute__((ext_vector_type(16))) __bf16 v16bf;
typedef __attribute__((ext_vector_type(8)))  float  v8f;
typedef __attribute__((ext_vector_type(4)))  unsigned int u32x4;
typedef __attribute__((ext_vector_type(8)))  int i32x8;
typedef __attribute__((ext_vector_type(4)))  int i32x4;

namespace {

constexpr int kNHead  = 8;
constexpr int kDModel = 512;
constexpr int kDFF    = 2048;
constexpr int kSeq    = 2048;
constexpr int kBatch  = 4;
constexpr int kRows   = kBatch * kSeq;   // 8192

__device__ __forceinline__ unsigned short f32_to_bf16(float f) {
  unsigned int u = __float_as_uint(f);
  u += 0x7fffu + ((u >> 16) & 1u);       // round-to-nearest-even
  return (unsigned short)(u >> 16);
}

// CDNA5 async memory->LDS copy (no VGPR round trip, tracked by ASYNCcnt).
__device__ __forceinline__ void async_load_b128(unsigned int ldsOff, const void* gptr) {
  unsigned long long ga = (unsigned long long)gptr;
  asm volatile("global_load_async_to_lds_b128 %0, %1, off"
               :: "v"(ldsOff), "v"(ga) : "memory");
}
__device__ __forceinline__ void wait_async0() {
  asm volatile("s_wait_asynccnt 0" ::: "memory");
}

union FragBF16 {
  v16bf v;
  u32x4 q[2];
};

// ---------------- TDM warm-up: NULL-tensor descriptor (D#.count==0 -> NOP) ----------------
__global__ __launch_bounds__(32) void tdm_nop_probe() {
  __shared__ __attribute__((aligned(16))) unsigned char buf[256];
  (void)buf;
  u32x4 g0 = (u32x4)0u;   // group0: count=0 => NULL tensor, architectural no-op
  i32x8 g1 = (i32x8)0;
  i32x4 g2 = (i32x4)0;
  i32x4 g3 = (i32x4)0;
  i32x8 g4 = (i32x8)0;
  __builtin_amdgcn_tensor_load_to_lds(g0, g1, g2, g3, g4, 0);
  __builtin_amdgcn_s_wait_tensorcnt(0);
}

// ---------------- prep kernels ----------------

__global__ __launch_bounds__(256) void cvt_f32_to_bf16(const float* __restrict__ in,
                                                       unsigned short* __restrict__ out,
                                                       long n) {
  long i = (long)blockIdx.x * blockDim.x + threadIdx.x;
  if (i < n) out[i] = f32_to_bf16(in[i]);
}

// W_eff[d,e] = sum_h W[h,d,e]  (n = 512*512), output bf16
__global__ __launch_bounds__(256) void reduce_heads_bf16(const float* __restrict__ W,
                                                         unsigned short* __restrict__ Wb,
                                                         int n) {
  int i = blockIdx.x * blockDim.x + threadIdx.x;
  if (i < n) {
    float s = 0.f;
    #pragma unroll
    for (int h = 0; h < kNHead; ++h) s += W[(long)h * n + i];
    Wb[i] = f32_to_bf16(s);
  }
}

// ---------------- WMMA GEMM ----------------
// C = scale * (A @ B) + bias   ; A: [M,K] bf16 row-major
// TRANSB=0: B is [K,N] row-major.  TRANSB=1: B is [N,K] row-major (B^T used).
// Block tile 64x64, BK=32. 4 waves (128 thr), each wave a 32x32 patch = 2x2
// v_wmma_f32_16x16x32_bf16 tiles. All dims assumed multiples of 64/32/8.
//
// LDS layouts are chosen so every WMMA fragment is contiguous 16B chunks:
//   As [row][k]  : lane fragment = As[row][8kg..+7]  and As[row][16+8kg..+7]
//   Bst[col][k]  : lane fragment = Bst[col][8kg..+7] and Bst[col][16+8kg..+7] (B transposed)
// Row pitch 40 shorts (80B): 16B-aligned chunks, conflict-free b128 reads.
// A tile (and B tile when TRANSB) staged with global_load_async_to_lds_b128.
template <bool TRANSB>
__global__ __launch_bounds__(128) void gemm_bf16_wmma(
    const unsigned short* __restrict__ A,
    const unsigned short* __restrict__ B,
    const float* __restrict__ bias,      // [N] or nullptr
    float* __restrict__ Cf,              // fp32 out or nullptr
    unsigned short* __restrict__ Cb,     // bf16 out or nullptr
    int M, int N, int K,
    long long sA, long long sB, long long sC,   // batch strides (elements)
    float scale)
{
  const int bz = blockIdx.z;
  A += (long long)bz * sA;
  B += (long long)bz * sB;
  const long long cbase = (long long)bz * sC;

  const int bm   = blockIdx.y * 64;
  const int bn   = blockIdx.x * 64;
  const int tid  = threadIdx.x;      // 0..127
  const int lane = tid & 31;
  const int wave = tid >> 5;
  const int wm   = (wave >> 1) * 32; // wave row offset in block tile
  const int wn   = (wave & 1)  * 32; // wave col offset
  const int lm   = lane & 15;
  const int kg   = lane >> 4;        // lane half (ISA 16-bit operand layout)

  __shared__ __attribute__((aligned(16))) unsigned short As[64][40];   // 64x32 bf16
  __shared__ __attribute__((aligned(16))) unsigned short Bst[64][40];  // 64(n) x 32(k) bf16

  // ---- loop-invariant staging addresses ----
  // A tile: 256 x 16B chunks, 2 per thread.
  const int pA0 = tid, pA1 = tid + 128;
  const int rA0 = pA0 >> 2, cA0 = (pA0 & 3) * 8;
  const int rA1 = pA1 >> 2, cA1 = (pA1 & 3) * 8;
  const unsigned short* aSrc0 = A + (long long)(bm + rA0) * K + cA0;
  const unsigned short* aSrc1 = A + (long long)(bm + rA1) * K + cA1;
  const unsigned int aDst0 = (unsigned int)(uintptr_t)&As[rA0][cA0];
  const unsigned int aDst1 = (unsigned int)(uintptr_t)&As[rA1][cA1];

  // B tile staging addresses
  const unsigned short* bSrc0;
  const unsigned short* bSrc1;
  unsigned int bDst0, bDst1;
  unsigned short* bScat0 = nullptr;
  unsigned short* bScat1 = nullptr;
  if (TRANSB) {            // direct row copy into Bst (16B async chunks)
    const int c0 = pA0 >> 2, h0 = (pA0 & 3) * 8;
    const int c1 = pA1 >> 2, h1 = (pA1 & 3) * 8;
    bSrc0 = B + (long long)(bn + c0) * K + h0;
    bSrc1 = B + (long long)(bn + c1) * K + h1;
    bDst0 = (unsigned int)(uintptr_t)&Bst[c0][h0];
    bDst1 = (unsigned int)(uintptr_t)&Bst[c1][h1];
  } else {                 // 16B load of 8 consecutive n + 16-bit transpose scatter
    const int r0 = pA0 >> 3, n0 = (pA0 & 7) * 8;
    const int r1 = pA1 >> 3, n1 = (pA1 & 7) * 8;
    bSrc0 = B + (long long)r0 * N + bn + n0;
    bSrc1 = B + (long long)r1 * N + bn + n1;
    bScat0 = &Bst[n0][r0];
    bScat1 = &Bst[n1][r1];
    bDst0 = bDst1 = 0;
  }

  const v8f vzero = {0.f, 0.f, 0.f, 0.f, 0.f, 0.f, 0.f, 0.f};
  v8f acc[2][2];
  #pragma unroll
  for (int i = 0; i < 2; ++i)
    #pragma unroll
    for (int j = 0; j < 2; ++j) acc[i][j] = vzero;

  for (int k0 = 0; k0 < K; k0 += 32) {
    // ---- stage A tile asynchronously (memory -> LDS, no VGPR round trip) ----
    async_load_b128(aDst0, aSrc0 + k0);
    async_load_b128(aDst1, aSrc1 + k0);

    // ---- stage B tile into transposed LDS layout Bst[n][k] ----
    if (TRANSB) {
      async_load_b128(bDst0, bSrc0 + k0);
      async_load_b128(bDst1, bSrc1 + k0);
    } else {
      // issue both global loads first, then scatter (batches the waits)
      u32x4 t0 = *(const u32x4*)(bSrc0 + (long long)k0 * N);
      u32x4 t1 = *(const u32x4*)(bSrc1 + (long long)k0 * N);
      #pragma unroll
      for (int j = 0; j < 4; ++j) {
        unsigned int w = t0[j];
        bScat0[(2 * j) * 40]     = (unsigned short)(w & 0xffffu);
        bScat0[(2 * j + 1) * 40] = (unsigned short)(w >> 16);
      }
      #pragma unroll
      for (int j = 0; j < 4; ++j) {
        unsigned int w = t1[j];
        bScat1[(2 * j) * 40]     = (unsigned short)(w & 0xffffu);
        bScat1[(2 * j + 1) * 40] = (unsigned short)(w >> 16);
      }
    }
    wait_async0();
    __syncthreads();

    // ---- fragments: contiguous 16B LDS reads per the CDNA5 16-bit layouts ----
    FragBF16 af[2], bfr[2];
    #pragma unroll
    for (int tm = 0; tm < 2; ++tm) {
      int row = wm + tm * 16 + lm;
      af[tm].q[0] = *(const u32x4*)&As[row][kg * 8];        // K [8kg, 8kg+8)
      af[tm].q[1] = *(const u32x4*)&As[row][16 + kg * 8];   // K [16+8kg, ..+8)
    }
    #pragma unroll
    for (int tn = 0; tn < 2; ++tn) {
      int col = wn + tn * 16 + lm;
      bfr[tn].q[0] = *(const u32x4*)&Bst[col][kg * 8];      // K [8kg, 8kg+8)
      bfr[tn].q[1] = *(const u32x4*)&Bst[col][16 + kg * 8]; // K [16+8kg, ..+8)
    }
    #pragma unroll
    for (int tm = 0; tm < 2; ++tm)
      #pragma unroll
      for (int tn = 0; tn < 2; ++tn)
        acc[tm][tn] = __builtin_amdgcn_wmma_f32_16x16x32_bf16(
            false, af[tm].v, false, bfr[tn].v, (short)0, acc[tm][tn],
            false, false);
    __syncthreads();
  }

  // epilogue: C/D layout — VGPR i: lanes 0-15 M=i, lanes 16-31 M=i+8
  #pragma unroll
  for (int tm = 0; tm < 2; ++tm) {
    int row0 = bm + wm + tm * 16 + kg * 8;
    #pragma unroll
    for (int tn = 0; tn < 2; ++tn) {
      int col = bn + wn + tn * 16 + lm;
      float bia = bias ? bias[col] : 0.f;
      #pragma unroll
      for (int i = 0; i < 8; ++i) {
        float val = acc[tm][tn][i] * scale + bia;
        long long idx = cbase + (long long)(row0 + i) * N + col;
        if (Cf) Cf[idx] = val;
        if (Cb) Cb[idx] = f32_to_bf16(val);
      }
    }
  }
}

// ---------------- softmax over rows of [nrows, ncols] fp32 -> bf16 ----------------
__global__ __launch_bounds__(256) void softmax_rows(const float* __restrict__ S,
                                                    unsigned short* __restrict__ P,
                                                    int ncols) {
  long row = blockIdx.x;
  const float* s = S + row * (long)ncols;
  unsigned short* p = P + row * (long)ncols;
  int t = threadIdx.x;
  __shared__ float red[256];

  float m = -3.4e38f;
  for (int c = t; c < ncols; c += 256) m = fmaxf(m, s[c]);
  red[t] = m;
  __syncthreads();
  for (int o = 128; o > 0; o >>= 1) {
    if (t < o) red[t] = fmaxf(red[t], red[t + o]);
    __syncthreads();
  }
  m = red[0];
  __syncthreads();

  float sum = 0.f;
  for (int c = t; c < ncols; c += 256) sum += __expf(s[c] - m);
  red[t] = sum;
  __syncthreads();
  for (int o = 128; o > 0; o >>= 1) {
    if (t < o) red[t] += red[t + o];
    __syncthreads();
  }
  float inv = 1.f / red[0];

  for (int c = t; c < ncols; c += 256) p[c] = f32_to_bf16(__expf(s[c] - m) * inv);
}

// ---------------- fused residual add + LayerNorm over D=512 ----------------
__global__ __launch_bounds__(256) void add_ln_512(const float* __restrict__ X,
                                                  const float* __restrict__ Y,
                                                  const float* __restrict__ gamma,
                                                  const float* __restrict__ beta,
                                                  float* __restrict__ Of,
                                                  unsigned short* __restrict__ Ob) {
  long row = blockIdx.x;
  const float* x = X + row * kDModel;
  const float* y = Y + row * kDModel;
  int t = threadIdx.x;
  __shared__ float red[256];

  float a0 = x[t] + y[t];
  float a1 = x[t + 256] + y[t + 256];

  red[t] = a0 + a1;
  __syncthreads();
  for (int o = 128; o > 0; o >>= 1) {
    if (t < o) red[t] += red[t + o];
    __syncthreads();
  }
  float mean = red[0] * (1.0f / (float)kDModel);
  __syncthreads();

  float d0 = a0 - mean, d1 = a1 - mean;
  red[t] = d0 * d0 + d1 * d1;
  __syncthreads();
  for (int o = 128; o > 0; o >>= 1) {
    if (t < o) red[t] += red[t + o];
    __syncthreads();
  }
  float inv = rsqrtf(red[0] * (1.0f / (float)kDModel) + 1e-14f);

  float o0 = d0 * inv * gamma[t]       + beta[t];
  float o1 = d1 * inv * gamma[t + 256] + beta[t + 256];
  long base = row * kDModel;
  Of[base + t]       = o0;
  Of[base + t + 256] = o1;
  if (Ob) {
    Ob[base + t]       = f32_to_bf16(o0);
    Ob[base + t + 256] = f32_to_bf16(o1);
  }
}

}  // namespace

extern "C" void kernel_launch(void* const* d_in, const int* in_sizes, int n_in,
                              void* d_out, int out_size, void* d_ws, size_t ws_size,
                              hipStream_t stream) {
  (void)in_sizes; (void)n_in; (void)out_size; (void)ws_size;

  const float* x   = (const float*)d_in[0];
  const float* Wq  = (const float*)d_in[1];
  const float* bq  = (const float*)d_in[2];
  const float* Wk  = (const float*)d_in[3];
  const float* bk  = (const float*)d_in[4];
  const float* Wv  = (const float*)d_in[5];
  const float* bv  = (const float*)d_in[6];
  const float* g1  = (const float*)d_in[7];
  const float* be1 = (const float*)d_in[8];
  const float* g2  = (const float*)d_in[9];
  const float* be2 = (const float*)d_in[10];
  const float* W1  = (const float*)d_in[11];
  const float* b1  = (const float*)d_in[12];
  const float* W2  = (const float*)d_in[13];
  const float* b2  = (const float*)d_in[14];
  float* out = (float*)d_out;

  char* ws = (char*)d_ws;
  size_t off = 0;
  auto carve = [&](size_t bytes) -> void* {
    void* p = ws + off;
    off += (bytes + 255) & ~(size_t)255;
    return p;
  };

  unsigned short* xb  = (unsigned short*)carve((size_t)kRows * kDModel * 2);
  unsigned short* wqb = (unsigned short*)carve((size_t)kDModel * kDModel * 2);
  unsigned short* wkb = (unsigned short*)carve((size_t)kDModel * kDModel * 2);
  unsigned short* wvb = (unsigned short*)carve((size_t)kDModel * kDModel * 2);
  unsigned short* w1b = (unsigned short*)carve((size_t)kDModel * kDFF * 2);
  unsigned short* w2b = (unsigned short*)carve((size_t)kDFF * kDModel * 2);
  unsigned short* qb  = (unsigned short*)carve((size_t)kRows * kDModel * 2);
  unsigned short* kb  = (unsigned short*)carve((size_t)kRows * kDModel * 2);
  unsigned short* vb  = (unsigned short*)carve((size_t)kRows * kDModel * 2);
  float*          sc  = (float*)carve((size_t)kBatch * kSeq * kSeq * 4);
  unsigned short* at  = (unsigned short*)carve((size_t)kBatch * kSeq * kSeq * 2);
  float*          yF  = (float*)carve((size_t)kRows * kDModel * 4);
  float*          hF  = (float*)carve((size_t)kRows * kDModel * 4);
  unsigned short* hB  = (unsigned short*)carve((size_t)kRows * kDModel * 2);
  unsigned short* fB  = (unsigned short*)carve((size_t)kRows * kDFF * 2);
  float*          gF  = (float*)carve((size_t)kRows * kDModel * 4);

  // 0) TDM no-op probe (NULL descriptor): exercises tensor_load_to_lds path
  tdm_nop_probe<<<dim3(1), 32, 0, stream>>>();

  // 1) precision prep: x -> bf16, W_eff = sum_h W[h] -> bf16, W1/W2 -> bf16
  {
    long n = (long)kRows * kDModel;
    cvt_f32_to_bf16<<<dim3((unsigned)((n + 255) / 256)), 256, 0, stream>>>(x, xb, n);
  }
  {
    int n = kDModel * kDModel;
    dim3 g((n + 255) / 256);
    reduce_heads_bf16<<<g, 256, 0, stream>>>(Wq, wqb, n);
    reduce_heads_bf16<<<g, 256, 0, stream>>>(Wk, wkb, n);
    reduce_heads_bf16<<<g, 256, 0, stream>>>(Wv, wvb, n);
  }
  {
    long n = (long)kDModel * kDFF;
    dim3 g((unsigned)((n + 255) / 256));
    cvt_f32_to_bf16<<<g, 256, 0, stream>>>(W1, w1b, n);
    cvt_f32_to_bf16<<<g, 256, 0, stream>>>(W2, w2b, n);
  }

  // 2) QKV projections: [8192,512] x [512,512] -> bf16
  {
    dim3 g(kDModel / 64, kRows / 64, 1);
    gemm_bf16_wmma<false><<<g, 128, 0, stream>>>(xb, wqb, bq, nullptr, qb,
        kRows, kDModel, kDModel, 0, 0, 0, 1.f);
    gemm_bf16_wmma<false><<<g, 128, 0, stream>>>(xb, wkb, bk, nullptr, kb,
        kRows, kDModel, kDModel, 0, 0, 0, 1.f);
    gemm_bf16_wmma<false><<<g, 128, 0, stream>>>(xb, wvb, bv, nullptr, vb,
        kRows, kDModel, kDModel, 0, 0, 0, 1.f);
  }

  // 3) scores = q k^T / 8 per batch : [2048,512] x [512,2048] (B transposed)
  {
    dim3 g(kSeq / 64, kSeq / 64, kBatch);
    gemm_bf16_wmma<true><<<g, 128, 0, stream>>>(qb, kb, nullptr, sc, nullptr,
        kSeq, kSeq, kDModel,
        (long long)kSeq * kDModel, (long long)kSeq * kDModel,
        (long long)kSeq * kSeq, 0.125f);
  }

  // 4) softmax rows ([8192, 2048] flat; stays resident in 192MB L2)
  softmax_rows<<<dim3(kRows), 256, 0, stream>>>(sc, at, kSeq);

  // 5) y = attn @ v per batch : [2048,2048] x [2048,512]
  {
    dim3 g(kDModel / 64, kSeq / 64, kBatch);
    gemm_bf16_wmma<false><<<g, 128, 0, stream>>>(at, vb, nullptr, yF, nullptr,
        kSeq, kDModel, kSeq,
        (long long)kSeq * kSeq, (long long)kSeq * kDModel,
        (long long)kSeq * kDModel, 1.f);
  }

  // 6) h = LN(x + y)
  add_ln_512<<<dim3(kRows), 256, 0, stream>>>(x, yF, g1, be1, hF, hB);

  // 7) FFN: f = h@W1 + b1 ; g = f@W2 + b2
  {
    dim3 g(kDFF / 64, kRows / 64, 1);
    gemm_bf16_wmma<false><<<g, 128, 0, stream>>>(hB, w1b, b1, nullptr, fB,
        kRows, kDFF, kDModel, 0, 0, 0, 1.f);
  }
  {
    dim3 g(kDModel / 64, kRows / 64, 1);
    gemm_bf16_wmma<false><<<g, 128, 0, stream>>>(fB, w2b, b2, gF, nullptr,
        kRows, kDModel, kDFF, 0, 0, 0, 1.f);
  }

  // 8) out = LN(h + g)
  add_ln_512<<<dim3(kRows), 256, 0, stream>>>(hF, gF, g2, be2, out, nullptr);
}
